// GCN_network_54391465836833
// MI455X (gfx1250) — matrix operand order repaired and checked by reading, not compiled, not run on hardware
//
#include <hip/hip_runtime.h>
#include <hip/hip_bf16.h>
#include <math.h>

typedef __attribute__((ext_vector_type(16))) _Float16 v16h;
typedef __attribute__((ext_vector_type(8)))  _Float16 v8h;
typedef __attribute__((ext_vector_type(8)))  float    v8f;

#define NN 327680      // nodes
#define NE 2621440     // edges

// ---------------------------------------------------------------- utilities
__global__ void k_fill_f32(float* __restrict__ p, float v, long n) {
  long i = (long)blockIdx.x * blockDim.x + threadIdx.x;
  if (i < n) p[i] = v;
}

__global__ void k_deg_edges(const int* __restrict__ dst, float* __restrict__ deg, int e) {
  int i = blockIdx.x * blockDim.x + threadIdx.x;
  if (i < e) atomicAdd(&deg[dst[i]], 1.0f);
}

__global__ void k_rsqrt_inplace(float* __restrict__ p, int n) {
  int i = blockIdx.x * blockDim.x + threadIdx.x;
  if (i < n) p[i] = rsqrtf(p[i]);   // deg >= 1 always (self loop)
}

// convert f32 [ri,ci] -> f16 [ro,co] row-major, zero padding extra rows/cols (A matrices)
__global__ void k_cvt_pad(const float* __restrict__ in, _Float16* __restrict__ out,
                          int ri, int ro, int ci, int co) {
  long i = (long)blockIdx.x * blockDim.x + threadIdx.x;
  long total = (long)ro * co;
  if (i >= total) return;
  int r = (int)(i / co), c = (int)(i % co);
  float v = (r < ri && c < ci) ? in[(long)r * ci + c] : 0.0f;
  out[i] = (_Float16)v;
}

// Pack f32 weight [Ki,Ci] into WMMA B-fragment order, f16, padded to [Kp,N]:
// out[((nt*ksteps + ks)*32 + lane)*16 + e] = B[ks*32 + (lane>>4)*16 + e, nt*16 + (lane&15)]
// so each lane's v16h fragment is 32 contiguous bytes.
__global__ void k_pack_b(const float* __restrict__ in, _Float16* __restrict__ out,
                         int Kp, int N, int Ki, int Ci) {
  long i = (long)blockIdx.x * blockDim.x + threadIdx.x;
  long total = (long)N * Kp;
  if (i >= total) return;
  int ksteps = Kp >> 5;
  int e    = (int)(i & 15);
  int lane = (int)((i >> 4) & 31);
  long t   = i >> 9;
  int ks   = (int)(t % ksteps);
  int nt   = (int)(t / ksteps);
  int k = ks * 32 + (lane >> 4) * 16 + e;
  int n = nt * 16 + (lane & 15);
  float v = (k < Ki && n < Ci) ? in[(long)k * Ci + n] : 0.0f;
  out[i] = (_Float16)v;
}

// ------------------------------------------------ WMMA GEMM (wave32, 16x16 tile)
// A [M,K] f16 row-major (K % 32 == 0); Bf fragment-packed f16 (see k_pack_b).
// C f32 out and/or Ch f16 out. One wave per 16x16 D tile; 4 waves/block over M.
// K loop is software-pipelined: WMMA(k) only depends on loads issued at k-1,
// so VMEM latency overlaps the matrix pipe within a single wave.
__global__ void k_wmma_gemm(const _Float16* __restrict__ A,
                            const _Float16* __restrict__ Bf,
                            const float*    __restrict__ bias,
                            float*          __restrict__ C,
                            _Float16*       __restrict__ Ch,
                            int M, int N, int K, int relu) {
  int wave = threadIdx.x >> 5;
  int lane = threadIdx.x & 31;
  int nt   = blockIdx.x;
  int row  = (blockIdx.y * 4 + wave) * 16;
  int col  = nt * 16;
  int m    = lane & 15;        // A-row / D-col within tile
  int half = lane >> 4;        // lane-half selector
  int ksteps = K >> 5;

  v8f acc = {};
  // A fragment: per lane, halfs [half*16 .. +15] (contiguous, matches k_pack_b)
  const _Float16* ap = A + (long)(row + m) * K + half * 16;
  const _Float16* bq = Bf + ((long)nt * ksteps * 32 + lane) * 16;

  union F { v16h v; v8h h[2]; };
  F a_cur, a_nxt;
  v16h b_cur, b_nxt;
  a_cur.h[0] = *(const v8h*)(ap);        // 16B aligned
  a_cur.h[1] = *(const v8h*)(ap + 8);
  b_cur      = *(const v16h*)(bq);       // 32B aligned, contiguous fragment

  for (int ks = 0; ks < ksteps - 1; ++ks) {
    ap += 32;
    bq += 512;
    a_nxt.h[0] = *(const v8h*)(ap);
    a_nxt.h[1] = *(const v8h*)(ap + 8);
    b_nxt      = *(const v16h*)(bq);
    __builtin_prefetch(ap + 32, 0, 1);   // global_prefetch_b8, 2 steps ahead
    acc = __builtin_amdgcn_wmma_f32_16x16x32_f16(false, a_cur.v, false, b_cur,
                                                 (short)0, acc, false, false);
    a_cur.v = a_nxt.v;
    b_cur   = b_nxt;
  }
  acc = __builtin_amdgcn_wmma_f32_16x16x32_f16(false, a_cur.v, false, b_cur,
                                               (short)0, acc, false, false);

  // D 16x16 f32: VGPR r -> M = r + 8*half, N = lane%16
#pragma unroll
  for (int r = 0; r < 8; ++r) {
    int orow = row + r + 8 * half;
    int ocol = col + m;
    float v = acc[r];
    if (bias) v += bias[ocol];
    if (relu) v = fmaxf(v, 0.0f);
    long idx = (long)orow * N + ocol;
    if (C)  C[idx]  = v;
    if (Ch) Ch[idx] = (_Float16)v;
  }
}

// ------------------------------------------- normalized edge scatter (atomics)
// edges e < E use (src[e],dst[e]); e >= E are self loops (node e-E).
// Each thread handles one float4 channel chunk of one edge.
__global__ void k_scatter(const int* __restrict__ src, const int* __restrict__ dst,
                          const float* __restrict__ dinv,
                          const float* __restrict__ h, float* __restrict__ agg,
                          int E, int n, int C4) {
  long i = (long)blockIdx.x * blockDim.x + threadIdx.x;
  long total = (long)(E + n) * C4;
  if (i >= total) return;
  int  c = (int)(i % C4);
  long e = i / C4;
  int s, d;
  if (e < E) { s = src[e]; d = dst[e]; } else { s = d = (int)(e - E); }
  float norm = dinv[s] * dinv[d];
  float4 v = *((const float4*)(h + (long)s * C4 * 4) + c);
  float* ap = agg + (long)d * C4 * 4 + (long)c * 4;
  atomicAdd(ap + 0, v.x * norm);
  atomicAdd(ap + 1, v.y * norm);
  atomicAdd(ap + 2, v.z * norm);
  atomicAdd(ap + 3, v.w * norm);
}

// bias + ReLU + fp16 convert
__global__ void k_postagg(const float* __restrict__ agg, const float* __restrict__ bias,
                          _Float16* __restrict__ outh, int n, int C) {
  long i = (long)blockIdx.x * blockDim.x + threadIdx.x;
  if (i >= (long)n * C) return;
  int c = (int)(i % C);
  outh[i] = (_Float16)fmaxf(agg[i] + bias[c], 0.0f);
}

// final [8192,64] @ [64,1] + sigmoid
__global__ void k_final(const float* __restrict__ g2, const float* __restrict__ w,
                        const float* __restrict__ b, float* __restrict__ out, int rows) {
  int i = blockIdx.x * blockDim.x + threadIdx.x;
  if (i >= rows) return;
  const float* r = g2 + (long)i * 64;
  float acc = b[0];
#pragma unroll
  for (int k = 0; k < 64; ++k) acc = fmaf(r[k], w[k], acc);
  out[i] = 1.0f / (1.0f + expf(-acc));
}

static inline unsigned gs(long n, int b) { return (unsigned)((n + b - 1) / b); }

extern "C" void kernel_launch(void* const* d_in, const int* in_sizes, int n_in,
                              void* d_out, int out_size, void* d_ws, size_t ws_size,
                              hipStream_t stream) {
  (void)in_sizes; (void)n_in; (void)out_size; (void)ws_size;
  const float* x   = (const float*)d_in[0];
  const int*   ei  = (const int*)d_in[1];
  const int*   src = ei;
  const int*   dst = ei + NE;
  const float* W1  = (const float*)d_in[2];
  const float* b1  = (const float*)d_in[3];
  const float* W2  = (const float*)d_in[4];
  const float* b2  = (const float*)d_in[5];
  const float* l1w = (const float*)d_in[6];
  const float* l1b = (const float*)d_in[7];
  const float* l2w = (const float*)d_in[8];
  const float* l2b = (const float*)d_in[9];
  const float* l3w = (const float*)d_in[10];
  const float* l3b = (const float*)d_in[11];
  float* out = (float*)d_out;

  // ---- workspace carving (256B aligned), with reuse
  char* ws = (char*)d_ws;
  size_t off = 0;
  auto carve = [&](size_t bytes) -> char* {
    char* p = ws + off;
    off += (bytes + 255) & ~(size_t)255;
    return p;
  };
  float*    dinv = (float*)   carve((size_t)NN * 4);
  _Float16* xh   = (_Float16*)carve((size_t)NN * 64 * 2);   // reused later as g [8192,1920] f16
  _Float16* w1b  = (_Float16*)carve((size_t)64 * 96 * 2);
  _Float16* w2b  = (_Float16*)carve((size_t)96 * 48 * 2);
  _Float16* l1bf = (_Float16*)carve((size_t)1920 * 256 * 2);
  _Float16* l2bf = (_Float16*)carve((size_t)256 * 64 * 2);
  float*    hw1  = (float*)   carve((size_t)NN * 96 * 4);   // reused as hw2 [NN,48]
  float*    agg1 = (float*)   carve((size_t)NN * 96 * 4);   // reused as agg2 [NN,48]
  _Float16* h1h  = (_Float16*)carve((size_t)NN * 96 * 2);
  _Float16* g1h  = (_Float16*)carve((size_t)8192 * 256 * 2);
  float*    g2   = (float*)   carve((size_t)8192 * 64 * 4);
  float*    hw2  = hw1;
  float*    agg2 = agg1;
  _Float16* gh   = xh;

  const int B = 256;

  // ---- degree / normalization: deg = 1 (self loop) + in-degree; dinv = rsqrt(deg)
  k_fill_f32<<<gs(NN, B), B, 0, stream>>>(dinv, 1.0f, NN);
  k_deg_edges<<<gs(NE, B), B, 0, stream>>>(dst, dinv, NE);
  k_rsqrt_inplace<<<gs(NN, B), B, 0, stream>>>(dinv, NN);

  // ---- fp32 -> fp16 staging: A padded row-major, B fragment-packed
  k_cvt_pad<<<gs((long)NN * 64, B), B, 0, stream>>>(x, xh, NN, NN, 58, 64);
  k_pack_b<<<gs(64 * 96,    B), B, 0, stream>>>(W1,  w1b,  64,   96,  58,  96);
  k_pack_b<<<gs(96 * 48,    B), B, 0, stream>>>(W2,  w2b,  96,   48,  96,  48);
  k_pack_b<<<gs(1920 * 256, B), B, 0, stream>>>(l1w, l1bf, 1920, 256, 1920, 256);
  k_pack_b<<<gs(256 * 64,   B), B, 0, stream>>>(l2w, l2bf, 256,  64,  256,  64);

  // ---- conv1: hw1 = x @ W1   (M=327680, N=96, K=64)
  {
    dim3 grid(96 / 16, NN / 16 / 4);
    k_wmma_gemm<<<grid, 128, 0, stream>>>(xh, w1b, nullptr, hw1, nullptr, NN, 96, 64, 0);
  }
  k_fill_f32<<<gs((long)NN * 96, B), B, 0, stream>>>(agg1, 0.0f, (long)NN * 96);
  k_scatter<<<gs((long)(NE + NN) * 24, B), B, 0, stream>>>(src, dst, dinv, hw1, agg1, NE, NN, 24);
  k_postagg<<<gs((long)NN * 96, B), B, 0, stream>>>(agg1, b1, h1h, NN, 96);

  // ---- conv2: hw2 = h1 @ W2  (M=327680, N=48, K=96)
  {
    dim3 grid(48 / 16, NN / 16 / 4);
    k_wmma_gemm<<<grid, 128, 0, stream>>>(h1h, w2b, nullptr, hw2, nullptr, NN, 48, 96, 0);
  }
  k_fill_f32<<<gs((long)NN * 48, B), B, 0, stream>>>(agg2, 0.0f, (long)NN * 48);
  k_scatter<<<gs((long)(NE + NN) * 12, B), B, 0, stream>>>(src, dst, dinv, hw2, agg2, NE, NN, 12);
  k_postagg<<<gs((long)NN * 48, B), B, 0, stream>>>(agg2, b2, gh, NN, 48);   // gh = [8192,1920] contiguous

  // ---- graph MLP1: g1 = relu(g @ l1_w + l1_b)  (M=8192, N=256, K=1920), f16 out
  {
    dim3 grid(256 / 16, 8192 / 16 / 4);
    k_wmma_gemm<<<grid, 128, 0, stream>>>(gh, l1bf, l1b, nullptr, g1h, 8192, 256, 1920, 1);
  }
  // ---- graph MLP2: g2 = relu(g1 @ l2_w + l2_b)  (M=8192, N=64, K=256), f32 out
  {
    dim3 grid(64 / 16, 8192 / 16 / 4);
    k_wmma_gemm<<<grid, 128, 0, stream>>>(g1h, l2bf, l2b, g2, nullptr, 8192, 64, 256, 1);
  }
  // ---- head: sigmoid(g2 @ l3_w + l3_b)
  k_final<<<gs(8192, B), B, 0, stream>>>(g2, l3w, l3b, out, 8192);
}
